// NllbMoeEncoderLayer_83494164234449
// MI455X (gfx1250) — matrix operand order repaired and checked by reading, not compiled, not run on hardware
//
#include <hip/hip_runtime.h>
#include <cstdint>

// ---------------- types ----------------
typedef __bf16 bhalf;
typedef bhalf vb16 __attribute__((ext_vector_type(16)));
typedef bhalf vb8  __attribute__((ext_vector_type(8)));
typedef float v8f  __attribute__((ext_vector_type(8)));

#define SHUF16(lo, hh) __builtin_shufflevector(lo, hh, 0,1,2,3,4,5,6,7,8,9,10,11,12,13,14,15)

#define D_MODEL  1024
#define N_HEADS  16
#define HEAD_DIM 64
#define FFN_DIM  4096
#define N_EXP    8
#define SEQ      512
#define BATCH    4
#define T_TOK    2048     // BATCH*SEQ
#define CAP      512      // ceil(0.25 * 2048)
#define LN_EPS   1e-5f
#define F32_EPS  1.1920929e-07f

__device__ inline v8f wmma_bf16(vb16 a, vb16 b, v8f c) {
    // D = A(16x32 bf16) * B(32x16 bf16) + C(16x16 f32)
    return __builtin_amdgcn_wmma_f32_16x16x32_bf16(false, a, false, b, (short)0, c, false, false);
}

__device__ inline vb16 ldA16(const bhalf* p) {   // A-fragment: two b128 at +0,+32B
    vb8 lo = *(const vb8*)p;
    vb8 hh = *(const vb8*)(p + 16);
    return SHUF16(lo, hh);
}
__device__ inline vb16 ldB16(const bhalf* p) {   // B-fragment: two b128 at +0,+32B
    vb8 lo = *(const vb8*)p;
    vb8 hh = *(const vb8*)(p + 16);
    return SHUF16(lo, hh);
}

// ---------------- transpose + fp32->bf16 convert: dst[c*R+r] = (bf16)src[r*C+c] ----------------
__global__ __launch_bounds__(256) void convT_kernel(
    const float* __restrict__ src, bhalf* __restrict__ dst,
    int R, int C, long srcBatch, long dstBatch)
{
    __shared__ float tile[16][17];
    long zs = (long)blockIdx.z * srcBatch;
    long zd = (long)blockIdx.z * dstBatch;
    int r0 = blockIdx.y * 16, c0 = blockIdx.x * 16;
    int tx = threadIdx.x & 15, ty = threadIdx.x >> 4;
    tile[ty][tx] = src[zs + (long)(r0 + ty) * C + (c0 + tx)];
    __syncthreads();
    dst[zd + (long)(c0 + ty) * R + (r0 + tx)] = (bhalf)tile[tx][ty];
}

// ---------------- LayerNorm: fp32 in -> optional fp32 out + bf16 out ----------------
__global__ __launch_bounds__(256) void ln_kernel(
    const float* __restrict__ x, const float* __restrict__ g, const float* __restrict__ b,
    float* __restrict__ yf, bhalf* __restrict__ ybf)
{
    __shared__ float red[256];
    int t = blockIdx.x, tid = threadIdx.x;
    const float* row = x + (long)t * D_MODEL;
    float s = 0.f;
    for (int i = tid; i < D_MODEL; i += 256) s += row[i];
    red[tid] = s; __syncthreads();
    for (int o = 128; o; o >>= 1) { if (tid < o) red[tid] += red[tid + o]; __syncthreads(); }
    float mu = red[0] * (1.0f / D_MODEL);
    __syncthreads();
    float v = 0.f;
    for (int i = tid; i < D_MODEL; i += 256) { float d = row[i] - mu; v += d * d; }
    red[tid] = v; __syncthreads();
    for (int o = 128; o; o >>= 1) { if (tid < o) red[tid] += red[tid + o]; __syncthreads(); }
    float inv = rsqrtf(red[0] * (1.0f / D_MODEL) + LN_EPS);
    for (int i = tid; i < D_MODEL; i += 256) {
        float val = (row[i] - mu) * inv * g[i] + b[i];
        if (yf) yf[(long)t * D_MODEL + i] = val;
        ybf[(long)t * D_MODEL + i] = (bhalf)val;
    }
}

// ---------------- generic bf16 WMMA GEMM (2x-unrolled ping-pong pipeline) ----------------
// C[m,n] = act( A[m,:K] * BT[n,:K] + bias[n] ) * scale (+ resid[m,n])
// A: bf16 row-major [M,K]; BT: bf16 row-major [N,K] (i.e. W^T); out fp32 or bf16.
// flags: bit0 = relu, bit1 = store bf16.  K must be a multiple of 64.
// Each wave computes a 32x64 tile (2 M-tiles x 4 N-tiles); block = 8 waves = 256x64.
__global__ __launch_bounds__(256) void gemm_bf16_kernel(
    const bhalf* __restrict__ A, const bhalf* __restrict__ BT,
    const float* __restrict__ bias, const float* __restrict__ resid,
    void* __restrict__ Cout, int M, int N, int K,
    long aB, long bB, long cB, long biasB, float scale, int flags)
{
    int z = blockIdx.z;
    const bhalf* Az = A + (long)z * aB;
    const bhalf* Bz = BT + (long)z * bB;
    const float* biasz = bias ? bias + (long)z * biasB : nullptr;

    int wave = threadIdx.x >> 5, lane = threadIdx.x & 31;
    int hi = lane >> 4, mr = lane & 15;
    int m0  = blockIdx.y * 256 + wave * 32;
    int n0b = blockIdx.x * 64;

    const bhalf* arow0 = Az + (long)(m0 + mr) * K + hi * 8;
    const bhalf* arow1 = arow0 + (long)16 * K;
    const bhalf* brow[4];
#pragma unroll
    for (int i = 0; i < 4; i++)
        brow[i] = Bz + (long)(n0b + i * 16 + mr) * K + hi * 8;

    v8f acc0[4] = {}, acc1[4] = {};

    // set X preload (k = 0)
    vb16 xa0 = ldA16(arow0), xa1 = ldA16(arow1);
    vb16 xb[4];
#pragma unroll
    for (int i = 0; i < 4; i++) xb[i] = ldB16(brow[i]);

    int k0 = 0;
    for (; k0 + 64 < K; k0 += 64) {
        // load set Y (k0+32) while X is consumed
        vb16 ya0 = ldA16(arow0 + k0 + 32), ya1 = ldA16(arow1 + k0 + 32);
        vb16 yb[4];
#pragma unroll
        for (int i = 0; i < 4; i++) yb[i] = ldB16(brow[i] + k0 + 32);
        if (k0 + 128 < K) {   // global_prefetch_b8 two 64-steps ahead
            __builtin_prefetch(arow0 + k0 + 128, 0, 1);
            __builtin_prefetch(brow[0] + k0 + 128, 0, 1);
            __builtin_prefetch(brow[2] + k0 + 128, 0, 1);
        }
#pragma unroll
        for (int i = 0; i < 4; i++) acc0[i] = wmma_bf16(xa0, xb[i], acc0[i]);
#pragma unroll
        for (int i = 0; i < 4; i++) acc1[i] = wmma_bf16(xa1, xb[i], acc1[i]);
        // load set X (k0+64) while Y is consumed
        xa0 = ldA16(arow0 + k0 + 64); xa1 = ldA16(arow1 + k0 + 64);
#pragma unroll
        for (int i = 0; i < 4; i++) xb[i] = ldB16(brow[i] + k0 + 64);
#pragma unroll
        for (int i = 0; i < 4; i++) acc0[i] = wmma_bf16(ya0, yb[i], acc0[i]);
#pragma unroll
        for (int i = 0; i < 4; i++) acc1[i] = wmma_bf16(ya1, yb[i], acc1[i]);
    }
    // tail: k0 == K-64 here
    {
        vb16 ya0 = ldA16(arow0 + k0 + 32), ya1 = ldA16(arow1 + k0 + 32);
        vb16 yb[4];
#pragma unroll
        for (int i = 0; i < 4; i++) yb[i] = ldB16(brow[i] + k0 + 32);
#pragma unroll
        for (int i = 0; i < 4; i++) acc0[i] = wmma_bf16(xa0, xb[i], acc0[i]);
#pragma unroll
        for (int i = 0; i < 4; i++) acc1[i] = wmma_bf16(xa1, xb[i], acc1[i]);
#pragma unroll
        for (int i = 0; i < 4; i++) acc0[i] = wmma_bf16(ya0, yb[i], acc0[i]);
#pragma unroll
        for (int i = 0; i < 4; i++) acc1[i] = wmma_bf16(ya1, yb[i], acc1[i]);
    }

    long cbase = (long)z * cB;
    float* cf = (float*)Cout;
    bhalf* cb = (bhalf*)Cout;
#pragma unroll
    for (int t = 0; t < 2; t++) {
        v8f* acc = t ? acc1 : acc0;
#pragma unroll
        for (int i = 0; i < 4; i++) {
            int n = n0b + i * 16 + mr;
            float bv = biasz ? biasz[n] : 0.f;
#pragma unroll
            for (int r = 0; r < 8; r++) {
                int m = m0 + t * 16 + r + 8 * hi;
                float v = acc[i][r] + bv;
                if (flags & 1) v = fmaxf(v, 0.f);
                v *= scale;
                long off = (long)m * N + n;
                if (resid) v += resid[off];
                if (flags & 2) cb[cbase + off] = (bhalf)v;
                else           cf[cbase + off] = v;
            }
        }
    }
}

// ---------------- V transpose: vt[b][h][d][s] = vtmp[(b*S+s)*D + h*64 + d] ----------------
__global__ __launch_bounds__(256) void vtrans_kernel(
    const float* __restrict__ vtmp, bhalf* __restrict__ vt)
{
    int b = blockIdx.z, h = blockIdx.y, s0 = blockIdx.x * 64;
    int sl = threadIdx.x & 63, dg = threadIdx.x >> 6;
#pragma unroll
    for (int dd = 0; dd < 16; dd++) {
        int d = dg * 16 + dd;
        float v = vtmp[(long)(b * SEQ + s0 + sl) * D_MODEL + h * HEAD_DIM + d];
        vt[((long)(b * N_HEADS + h) * HEAD_DIM + d) * SEQ + s0 + sl] = (bhalf)v;
    }
}

// ---------------- attention: per (b, h, 64-row q-block); scores+softmax+PV in LDS ----------------
// LDS: 64x512 fp32 scores (128KB) + 64x512 bf16 probs (64KB) = 192KB dynamic.
__global__ __launch_bounds__(256) void attn_kernel(
    const bhalf* __restrict__ qbf, const bhalf* __restrict__ kbf,
    const bhalf* __restrict__ vt, bhalf* __restrict__ obf)
{
    extern __shared__ char smem[];
    float* sc = (float*)smem;                       // [64][512]
    bhalf* pr = (bhalf*)(smem + 64 * 512 * 4);      // [64][512]

    int b = blockIdx.z, h = blockIdx.y, q0 = blockIdx.x * 64;
    int wave = threadIdx.x >> 5, lane = threadIdx.x & 31;
    int hi = lane >> 4, mr = lane & 15;
    long headoff = (long)b * SEQ * D_MODEL + h * HEAD_DIM;

    int mt = wave >> 1;
    int m0 = mt * 16;

    // ---- phase 1: scores = Q @ K^T (q already pre-scaled by 1/8)
    vb16 qa0 = ldA16(qbf + headoff + (long)(q0 + m0 + mr) * D_MODEL + hi * 8);
    vb16 qa1 = ldA16(qbf + headoff + (long)(q0 + m0 + mr) * D_MODEL + 32 + hi * 8);

    const bhalf* kbase = kbf + headoff + (long)mr * D_MODEL + hi * 8;
    int nfirst = (wave & 1) * 256;   // first key column for this wave
    int nlast  = nfirst + 240;       // last valid tile start (clamp for lookahead)

    vb16 xk0 = ldB16(kbase + (long)nfirst * D_MODEL);
    vb16 xk1 = ldB16(kbase + (long)nfirst * D_MODEL + 32);
    for (int i = 0; i < 16; i += 2) {
        int n0 = nfirst + i * 16;
        // set Y = tile i+1
        const bhalf* kp1 = kbase + (long)(n0 + 16) * D_MODEL;
        vb16 yk0 = ldB16(kp1);
        vb16 yk1 = ldB16(kp1 + 32);
        v8f c = {};
        c = wmma_bf16(qa0, xk0, c);
        c = wmma_bf16(qa1, xk1, c);
#pragma unroll
        for (int r = 0; r < 8; r++)
            sc[(m0 + r + 8 * hi) * 512 + n0 + mr] = c[r];
        // set X = tile i+2 (clamped branchless lookahead)
        int nx = n0 + 32; if (nx > nlast) nx = nlast;
        const bhalf* kp2 = kbase + (long)nx * D_MODEL;
        xk0 = ldB16(kp2);
        xk1 = ldB16(kp2 + 32);
        v8f c2 = {};
        c2 = wmma_bf16(qa0, yk0, c2);
        c2 = wmma_bf16(qa1, yk1, c2);
#pragma unroll
        for (int r = 0; r < 8; r++)
            sc[(m0 + r + 8 * hi) * 512 + n0 + 16 + mr] = c2[r];
    }
    __syncthreads();

    // ---- phase 2: row softmax (wave handles 8 rows, 16 elems/lane)
    for (int rr = 0; rr < 8; rr++) {
        int row = wave * 8 + rr;
        float* srow = sc + row * 512;
        float vals[16], mx = -1e30f;
#pragma unroll
        for (int i = 0; i < 16; i++) { vals[i] = srow[lane + i * 32]; mx = fmaxf(mx, vals[i]); }
        for (int off = 16; off; off >>= 1) mx = fmaxf(mx, __shfl_xor(mx, off, 32));
        float sum = 0.f;
#pragma unroll
        for (int i = 0; i < 16; i++) { vals[i] = __expf(vals[i] - mx); sum += vals[i]; }
        for (int off = 16; off; off >>= 1) sum += __shfl_xor(sum, off, 32);
        float inv = 1.0f / sum;
        bhalf* prow = pr + row * 512;
#pragma unroll
        for (int i = 0; i < 16; i++) prow[lane + i * 32] = (bhalf)(vals[i] * inv);
    }
    __syncthreads();

    // ---- phase 3: O = P @ V (P from LDS, V pre-transposed per head)
    const bhalf* prow = pr + (m0 + mr) * 512 + hi * 8;
    for (int nn = 0; nn < 2; nn++) {
        int n0 = ((wave & 1) * 2 + nn) * 16;
        const bhalf* vrow = vt + ((long)(b * N_HEADS + h) * HEAD_DIM + n0 + mr) * SEQ + hi * 8;
        v8f c = {};
        vb16 xp = ldA16(prow);
        vb16 xv = ldB16(vrow);
        int k0 = 0;
        for (; k0 + 64 < 512; k0 += 64) {
            vb16 yp = ldA16(prow + k0 + 32);
            vb16 yv = ldB16(vrow + k0 + 32);
            c = wmma_bf16(xp, xv, c);
            xp = ldA16(prow + k0 + 64);
            xv = ldB16(vrow + k0 + 64);
            c = wmma_bf16(yp, yv, c);
        }
        vb16 yp = ldA16(prow + k0 + 32);
        vb16 yv = ldB16(vrow + k0 + 32);
        c = wmma_bf16(xp, xv, c);
        c = wmma_bf16(yp, yv, c);
#pragma unroll
        for (int r = 0; r < 8; r++) {
            int m = q0 + m0 + r + 8 * hi;
            obf[headoff + (long)m * D_MODEL + n0 + mr] = (bhalf)c[r];
        }
    }
}

// ---------------- router logits = h2 @ wr ----------------
__global__ __launch_bounds__(256) void router_kernel(
    const float* __restrict__ h2, const float* __restrict__ wr, float* __restrict__ logits)
{
    int idx = blockIdx.x * 256 + threadIdx.x;
    if (idx >= T_TOK * N_EXP) return;
    int t = idx >> 3, e = idx & 7;
    const float* row = h2 + (long)t * D_MODEL;
    float acc = 0.f;
    for (int i = 0; i < D_MODEL; i++) acc += row[i] * wr[i * N_EXP + e];
    logits[idx] = acc;
}

// ---------------- per-token softmax + top1/top2 ----------------
__global__ __launch_bounds__(256) void top2_kernel(
    const float* __restrict__ logits, int* __restrict__ e1, int* __restrict__ e2,
    float* __restrict__ p1, float* __restrict__ p2)
{
    int t = blockIdx.x * 256 + threadIdx.x;
    if (t >= T_TOK) return;
    float l[N_EXP], mx = -1e30f;
#pragma unroll
    for (int e = 0; e < N_EXP; e++) { l[e] = logits[t * N_EXP + e]; mx = fmaxf(mx, l[e]); }
    float pr[N_EXP], sum = 0.f;
#pragma unroll
    for (int e = 0; e < N_EXP; e++) { pr[e] = __expf(l[e] - mx); sum += pr[e]; }
    float inv = 1.f / sum;
    int a1 = 0; float b1v = pr[0];
#pragma unroll
    for (int e = 1; e < N_EXP; e++) if (pr[e] > b1v) { b1v = pr[e]; a1 = e; }
    int a2 = -1; float b2v = -1e30f;
#pragma unroll
    for (int e = 0; e < N_EXP; e++) if (e != a1 && l[e] > b2v) { b2v = l[e]; a2 = e; }
    e1[t] = a1; e2[t] = a2; p1[t] = pr[a1] * inv; p2[t] = pr[a2] * inv;
}

// ---------------- single-block capacity scan (exact cumsum semantics) ----------------
__global__ __launch_bounds__(256) void scan_kernel(
    const int* __restrict__ e1, const int* __restrict__ e2,
    const float* __restrict__ p1, const float* __restrict__ p2,
    int* __restrict__ s1, int* __restrict__ s2,
    float* __restrict__ g1, float* __restrict__ g2)
{
    __shared__ int c1[256][N_EXP], c2[256][N_EXP];
    __shared__ int tot1[N_EXP];
    int tid = threadIdx.x, t0 = tid * 8;
    int lc1[N_EXP] = {}, lc2[N_EXP] = {};
#pragma unroll
    for (int i = 0; i < 8; i++) { lc1[e1[t0 + i]]++; lc2[e2[t0 + i]]++; }
    for (int e = 0; e < N_EXP; e++) { c1[tid][e] = lc1[e]; c2[tid][e] = lc2[e]; }
    __syncthreads();
    if (tid < N_EXP) {
        int run = 0;
        for (int ch = 0; ch < 256; ch++) { int tmp = c1[ch][tid]; c1[ch][tid] = run; run += tmp; }
        tot1[tid] = run;   // total top1 count per expert (pre-capacity), offsets loc2
        run = 0;
        for (int ch = 0; ch < 256; ch++) { int tmp = c2[ch][tid]; c2[ch][tid] = run; run += tmp; }
    }
    __syncthreads();
    for (int e = 0; e < N_EXP; e++) { lc1[e] = 0; lc2[e] = 0; }
    for (int i = 0; i < 8; i++) {
        int t = t0 + i;
        int a = e1[t], b = e2[t];
        int sl1 = c1[tid][a] + lc1[a]++;
        int sl2 = c2[tid][b] + lc2[b]++ + tot1[b];
        bool v1 = sl1 < CAP, v2 = sl2 < CAP;
        float t1 = v1 ? p1[t] : 0.f;
        float t2 = v2 ? p2[t] : 0.f;
        float den = fmaxf(t1 + t2, F32_EPS);
        s1[t] = v1 ? sl1 : -1;
        s2[t] = v2 ? sl2 : -1;
        g1[t] = t1 / den;
        g2[t] = t2 / den;
    }
}

// ---------------- zero fill (u32) ----------------
__global__ __launch_bounds__(256) void zero_kernel(unsigned* __restrict__ p, long n)
{
    long i = (long)blockIdx.x * 256 + threadIdx.x;
    if (i < n) p[i] = 0u;
}

// ---------------- scatter tokens into capacity buffers ----------------
__global__ __launch_bounds__(256) void dispatch_kernel(
    const bhalf* __restrict__ h2bf, const int* __restrict__ e1, const int* __restrict__ e2,
    const int* __restrict__ s1, const int* __restrict__ s2, bhalf* __restrict__ buf)
{
    int t = blockIdx.x, tid = threadIdx.x;
    const unsigned* src = (const unsigned*)(h2bf + (long)t * D_MODEL);
    int a = s1[t];
    if (a >= 0) {
        unsigned* dst = (unsigned*)(buf + ((long)e1[t] * CAP + a) * D_MODEL);
        for (int i = tid; i < D_MODEL / 2; i += 256) dst[i] = src[i];
    }
    int b = s2[t];
    if (b >= 0) {
        unsigned* dst = (unsigned*)(buf + ((long)e2[t] * CAP + b) * D_MODEL);
        for (int i = tid; i < D_MODEL / 2; i += 256) dst[i] = src[i];
    }
}

// ---------------- gated combine + residual ----------------
__global__ __launch_bounds__(256) void combine_kernel(
    const float* __restrict__ x1, const float* __restrict__ y,
    const int* __restrict__ e1, const int* __restrict__ e2,
    const int* __restrict__ s1, const int* __restrict__ s2,
    const float* __restrict__ g1, const float* __restrict__ g2,
    float* __restrict__ out)
{
    int t = blockIdx.x, tid = threadIdx.x;
    int a = s1[t], b = s2[t];
    float ga = g1[t], gb = g2[t];
    const float* ya = (a >= 0) ? y + ((long)e1[t] * CAP + a) * D_MODEL : nullptr;
    const float* yb = (b >= 0) ? y + ((long)e2[t] * CAP + b) * D_MODEL : nullptr;
    for (int i = tid; i < D_MODEL; i += 256) {
        float v = x1[(long)t * D_MODEL + i];
        if (ya) v += ga * ya[i];
        if (yb) v += gb * yb[i];
        out[(long)t * D_MODEL + i] = v;
    }
}

// ================= host launcher =================
extern "C" void kernel_launch(void* const* d_in, const int* in_sizes, int n_in,
                              void* d_out, int out_size, void* d_ws, size_t ws_size,
                              hipStream_t stream)
{
    (void)in_sizes; (void)n_in; (void)out_size; (void)ws_size;
    const float* x    = (const float*)d_in[0];
    const float* wq   = (const float*)d_in[1];
    const float* bq   = (const float*)d_in[2];
    const float* wk   = (const float*)d_in[3];
    const float* bk   = (const float*)d_in[4];
    const float* wv   = (const float*)d_in[5];
    const float* bv   = (const float*)d_in[6];
    const float* wo   = (const float*)d_in[7];
    const float* bo   = (const float*)d_in[8];
    const float* ln1g = (const float*)d_in[9];
    const float* ln1b = (const float*)d_in[10];
    const float* ln2g = (const float*)d_in[11];
    const float* ln2b = (const float*)d_in[12];
    const float* wr   = (const float*)d_in[13];
    const float* w1   = (const float*)d_in[14];
    const float* b1   = (const float*)d_in[15];
    const float* w2   = (const float*)d_in[16];
    const float* b2   = (const float*)d_in[17];
    float* out = (float*)d_out;

    // bump allocator over workspace (~240 MB total)
    char* p = (char*)d_ws;
    auto alloc = [&](size_t bytes) -> char* {
        char* q = p; p += (bytes + 255) & ~(size_t)255; return q;
    };
    bhalf* wqT  = (bhalf*)alloc((size_t)D_MODEL * D_MODEL * 2);
    bhalf* wkT  = (bhalf*)alloc((size_t)D_MODEL * D_MODEL * 2);
    bhalf* wvT  = (bhalf*)alloc((size_t)D_MODEL * D_MODEL * 2);
    bhalf* woT  = (bhalf*)alloc((size_t)D_MODEL * D_MODEL * 2);
    bhalf* w1T  = (bhalf*)alloc((size_t)N_EXP * FFN_DIM * D_MODEL * 2);
    bhalf* w2T  = (bhalf*)alloc((size_t)N_EXP * FFN_DIM * D_MODEL * 2);
    bhalf* h1bf = (bhalf*)alloc((size_t)T_TOK * D_MODEL * 2);
    bhalf* qbf  = (bhalf*)alloc((size_t)T_TOK * D_MODEL * 2);
    bhalf* kbf  = (bhalf*)alloc((size_t)T_TOK * D_MODEL * 2);
    float* vtmp = (float*)alloc((size_t)T_TOK * D_MODEL * 4);
    bhalf* vt   = (bhalf*)alloc((size_t)T_TOK * D_MODEL * 2);
    bhalf* obf  = (bhalf*)alloc((size_t)T_TOK * D_MODEL * 2);
    float* x1   = (float*)alloc((size_t)T_TOK * D_MODEL * 4);
    float* h2   = (float*)alloc((size_t)T_TOK * D_MODEL * 4);
    bhalf* h2bf = (bhalf*)alloc((size_t)T_TOK * D_MODEL * 2);
    float* lg   = (float*)alloc((size_t)T_TOK * N_EXP * 4);
    int*   e1   = (int*)alloc((size_t)T_TOK * 4);
    int*   e2   = (int*)alloc((size_t)T_TOK * 4);
    float* p1   = (float*)alloc((size_t)T_TOK * 4);
    float* p2   = (float*)alloc((size_t)T_TOK * 4);
    int*   s1   = (int*)alloc((size_t)T_TOK * 4);
    int*   s2   = (int*)alloc((size_t)T_TOK * 4);
    float* g1   = (float*)alloc((size_t)T_TOK * 4);
    float* g2   = (float*)alloc((size_t)T_TOK * 4);
    bhalf* buf  = (bhalf*)alloc((size_t)N_EXP * CAP * D_MODEL * 2);
    bhalf* hexp = (bhalf*)alloc((size_t)N_EXP * CAP * FFN_DIM * 2);
    float* yexp = (float*)alloc((size_t)N_EXP * CAP * D_MODEL * 4);

    // --- weight transpose-convert to bf16 (B fragments want W^T row-major)
    convT_kernel<<<dim3(64, 64, 1), 256, 0, stream>>>(wq, wqT, D_MODEL, D_MODEL, 0, 0);
    convT_kernel<<<dim3(64, 64, 1), 256, 0, stream>>>(wk, wkT, D_MODEL, D_MODEL, 0, 0);
    convT_kernel<<<dim3(64, 64, 1), 256, 0, stream>>>(wv, wvT, D_MODEL, D_MODEL, 0, 0);
    convT_kernel<<<dim3(64, 64, 1), 256, 0, stream>>>(wo, woT, D_MODEL, D_MODEL, 0, 0);
    convT_kernel<<<dim3(FFN_DIM / 16, D_MODEL / 16, N_EXP), 256, 0, stream>>>(
        w1, w1T, D_MODEL, FFN_DIM, (long)D_MODEL * FFN_DIM, (long)FFN_DIM * D_MODEL);
    convT_kernel<<<dim3(D_MODEL / 16, FFN_DIM / 16, N_EXP), 256, 0, stream>>>(
        w2, w2T, FFN_DIM, D_MODEL, (long)FFN_DIM * D_MODEL, (long)D_MODEL * FFN_DIM);

    // --- LN1 (bf16 activations for QKV GEMMs)
    ln_kernel<<<T_TOK, 256, 0, stream>>>(x, ln1g, ln1b, nullptr, h1bf);

    // --- QKV projections (q pre-scaled by HEAD_DIM^-0.5); M-blocks of 256 rows
    gemm_bf16_kernel<<<dim3(16, 8, 1), 256, 0, stream>>>(
        h1bf, wqT, bq, nullptr, qbf, T_TOK, D_MODEL, D_MODEL, 0, 0, 0, 0, 0.125f, 2);
    gemm_bf16_kernel<<<dim3(16, 8, 1), 256, 0, stream>>>(
        h1bf, wkT, bk, nullptr, kbf, T_TOK, D_MODEL, D_MODEL, 0, 0, 0, 0, 1.0f, 2);
    gemm_bf16_kernel<<<dim3(16, 8, 1), 256, 0, stream>>>(
        h1bf, wvT, bv, nullptr, vtmp, T_TOK, D_MODEL, D_MODEL, 0, 0, 0, 0, 1.0f, 0);

    // --- transpose V per head for contiguous P@V B-fragments
    vtrans_kernel<<<dim3(SEQ / 64, N_HEADS, BATCH), 256, 0, stream>>>(vtmp, vt);

    // --- attention (192KB dynamic LDS per workgroup; CDNA5 allows 320KB)
    hipFuncSetAttribute((const void*)attn_kernel,
                        hipFuncAttributeMaxDynamicSharedMemorySize, 196608);
    attn_kernel<<<dim3(SEQ / 64, N_HEADS, BATCH), 256, 196608, stream>>>(qbf, kbf, vt, obf);

    // --- O projection + residual: x1 = x + (O @ wo + bo)
    gemm_bf16_kernel<<<dim3(16, 8, 1), 256, 0, stream>>>(
        obf, woT, bo, x, x1, T_TOK, D_MODEL, D_MODEL, 0, 0, 0, 0, 1.0f, 0);

    // --- LN2 (fp32 for router, bf16 for dispatch)
    ln_kernel<<<T_TOK, 256, 0, stream>>>(x1, ln2g, ln2b, h2, h2bf);

    // --- routing
    router_kernel<<<(T_TOK * N_EXP) / 256, 256, 0, stream>>>(h2, wr, lg);
    top2_kernel<<<T_TOK / 256, 256, 0, stream>>>(lg, e1, e2, p1, p2);
    scan_kernel<<<1, 256, 0, stream>>>(e1, e2, p1, p2, s1, s2, g1, g2);

    // --- dispatch into zeroed capacity buffers
    {
        long nwords = (long)N_EXP * CAP * D_MODEL / 2;
        zero_kernel<<<(unsigned)((nwords + 255) / 256), 256, 0, stream>>>((unsigned*)buf, nwords);
    }
    dispatch_kernel<<<T_TOK, 256, 0, stream>>>(h2bf, e1, e2, s1, s2, buf);

    // --- expert FFN: h = relu(buf @ w1 + b1) ; y = (h @ w2 + b2) * 0.8
    gemm_bf16_kernel<<<dim3(FFN_DIM / 64, CAP / 256, N_EXP), 256, 0, stream>>>(
        buf, w1T, b1, nullptr, hexp, CAP, FFN_DIM, D_MODEL,
        (long)CAP * D_MODEL, (long)FFN_DIM * D_MODEL, (long)CAP * FFN_DIM, FFN_DIM, 1.0f, 3);
    gemm_bf16_kernel<<<dim3(D_MODEL / 64, CAP / 256, N_EXP), 256, 0, stream>>>(
        hexp, w2T, b2, nullptr, yexp, CAP, D_MODEL, FFN_DIM,
        (long)CAP * FFN_DIM, (long)D_MODEL * FFN_DIM, (long)CAP * D_MODEL, D_MODEL, 0.8f, 0);

    // --- gated combine + residual -> out
    combine_kernel<<<T_TOK, 256, 0, stream>>>(x1, yexp, e1, e2, s1, s2, g1, g2, out);
}
